// Attention_73469710565477
// MI455X (gfx1250) — matrix operand order repaired and checked by reading, not compiled, not run on hardware
//
#include <hip/hip_runtime.h>
#include <hip/hip_bf16.h>
#include <math.h>

typedef __attribute__((ext_vector_type(16))) _Float16 v16h;
typedef __attribute__((ext_vector_type(8)))  _Float16 v8h;
typedef __attribute__((ext_vector_type(8)))  float    v8f;
typedef __attribute__((ext_vector_type(4)))  float    v4f;
typedef int v4i_vs __attribute__((vector_size(16)));   // matches builtin param type

#define NB_HEAD 16
#define DH      64
#define DM      1024
#define BB      4
#define SS      2048

#if defined(__gfx1250__) && __has_builtin(__builtin_amdgcn_global_load_async_to_lds_b128) && __has_builtin(__builtin_amdgcn_s_wait_asynccnt)
#define USE_ASYNC_LDS 1
#else
#define USE_ASYNC_LDS 0
#endif

static __device__ __forceinline__ v8f wmma_f16(v16h a, v16h b, v8f c) {
  // 8 args: (neg_a, A, neg_b, B, c_mod, C, reuse_a, reuse_b)
  return __builtin_amdgcn_wmma_f32_16x16x32_f16(false, a, false, b, (short)0, c, false, false);
}

// 16-byte global -> LDS copy. Async (ASYNCcnt) when the gfx1250 builtins are
// available; falls back to a synchronous register round-trip otherwise.
static __device__ __forceinline__ void copy16_g2l(const _Float16* g, _Float16* l) {
#if USE_ASYNC_LDS
  // Param types per hipcc diagnostic: AS(1) ("__device__") int4* source,
  // LDS destination. AS(1) ptr = full 64-bit address; AS(3) ptr = low 32 bits
  // of the generic LDS address.
  __builtin_amdgcn_global_load_async_to_lds_b128(
      (__attribute__((address_space(1))) v4i_vs*)(unsigned long long)(size_t)g,
      (__attribute__((address_space(3))) v4i_vs*)(unsigned int)(size_t)l,
      0, 0);
#else
  *(v8h*)l = *(const v8h*)g;
#endif
}

static __device__ __forceinline__ void wait_async_stage() {
#if USE_ASYNC_LDS
  __builtin_amdgcn_s_wait_asynccnt(0);
#endif
}

// ---------------------------------------------------------------------------
// Pass 0: W[1024,1024] f32 -> Wt[out][in] f16 (transpose + convert)
// ---------------------------------------------------------------------------
__global__ void wconv_kernel(const float* __restrict__ W, _Float16* __restrict__ Wt) {
  int idx = blockIdx.x * blockDim.x + threadIdx.x;   // 0 .. 1024*1024-1
  int o = idx & (DM - 1);
  int i = idx >> 10;
  Wt[(size_t)o * DM + i] = (_Float16)W[(size_t)i * DM + o];
}

// ---------------------------------------------------------------------------
// Pass 1: projection GEMM  X[B*S, 1024]f32 @ W[1024,1024] -> f16 per-head
// out_mode 0/1: [B,H,S,Dh] (q gets scale=1/8) ; out_mode 2: Vt [B,H,Dh,S]
// One wave = 16 rows x 64 cols (one head), K=1024 in chunks of 32.
// ---------------------------------------------------------------------------
__global__ __launch_bounds__(128) void proj_kernel(const float* __restrict__ X,
                                                   const _Float16* __restrict__ Wt,
                                                   _Float16* __restrict__ Out,
                                                   int out_mode, float scale) {
  int wave = (blockIdx.x * blockDim.x + threadIdx.x) >> 5;   // 0..8191
  int lane = threadIdx.x & 31;
  int head    = wave & (NB_HEAD - 1);
  int rowTile = wave >> 4;                                   // 0..511
  int row0 = rowTile * 16;                                   // row in [B*S]
  int b  = row0 / SS;
  int s0 = row0 % SS;
  int m    = lane & 15;
  int half = lane >> 4;

  const float*    xrow  = X  + (size_t)(row0 + m) * DM;
  const _Float16* wbase = Wt + (size_t)(head * DH) * DM;     // Wt rows = out cols

  v8f acc[4] = {};
  for (int kc = 0; kc < DM / 32; ++kc) {
    int d0 = kc * 32;
    __builtin_prefetch(xrow + d0 + 32, 0, 0);                // global_prefetch_b8
    // A fragment: f32 -> f16, K layout {16h..16h+15}
    v4f x0 = *(const v4f*)(xrow + d0 + 16 * half);
    v4f x1 = *(const v4f*)(xrow + d0 + 16 * half + 4);
    v4f x2 = *(const v4f*)(xrow + d0 + 16 * half + 8);
    v4f x3 = *(const v4f*)(xrow + d0 + 16 * half + 12);
    v16h a;
#pragma unroll
    for (int i = 0; i < 4; ++i) {
      a[i]      = (_Float16)x0[i];
      a[4 + i]  = (_Float16)x1[i];
      a[8 + i]  = (_Float16)x2[i];
      a[12 + i] = (_Float16)x3[i];
    }
#pragma unroll
    for (int nt = 0; nt < 4; ++nt) {
      const _Float16* wrow = wbase + (size_t)(nt * 16 + m) * DM + d0 + 16 * half;
      v8h b0 = *(const v8h*)(wrow);
      v8h b1 = *(const v8h*)(wrow + 8);
      v16h bf;
#pragma unroll
      for (int i = 0; i < 8; ++i) { bf[i] = b0[i]; bf[8 + i] = b1[i]; }
      acc[nt] = wmma_f16(a, bf, acc[nt]);
    }
  }
  // epilogue: C/D layout -> element (mm, n): VGPR mm%8, lane 16*(mm/8)+n
#pragma unroll
  for (int nt = 0; nt < 4; ++nt) {
#pragma unroll
    for (int r = 0; r < 8; ++r) {
      int mm = r + 8 * half;
      int s  = s0 + mm;
      int d  = nt * 16 + m;
      _Float16 val = (_Float16)(acc[nt][r] * scale);
      size_t addr;
      if (out_mode == 2)   // Vt [B,H,Dh,S]
        addr = (((size_t)b * NB_HEAD + head) * DH + d) * SS + s;
      else                 // [B,H,S,Dh]
        addr = (((size_t)b * NB_HEAD + head) * SS + s) * DH + d;
      Out[addr] = val;
    }
  }
}

// ---------------------------------------------------------------------------
// Pass 2: flash attention. One block = one (b,h,qgroup), 4 waves = 4 q-tiles of 16.
// K/V tiles (32 keys) staged to LDS asynchronously, double-buffered, shared
// by all 4 waves of the block.
// ---------------------------------------------------------------------------
__global__ __launch_bounds__(128) void attn_kernel(const _Float16* __restrict__ Qh,
                                                   const _Float16* __restrict__ Kh,
                                                   const _Float16* __restrict__ Vt,
                                                   const int* __restrict__ Qlen,
                                                   const int* __restrict__ Vlen,
                                                   float* __restrict__ Out) {
  __shared__ __align__(16) _Float16 kbuf[2][32 * 64];   // 2 x 4KB, [key][d]
  __shared__ __align__(16) _Float16 vbuf[2][64 * 32];   // 2 x 4KB, [d][key]
  __shared__ __align__(16) _Float16 pbuf[4][16 * 32];   // per-wave P tile

  int tid   = threadIdx.x;
  int wslot = tid >> 5;
  int lane  = tid & 31;
  int m     = lane & 15;
  int half  = lane >> 4;
  // 2048 blocks = 4b x 16h x 32 qgroups(of 4 tiles)
  int blk    = blockIdx.x;
  int qgroup = blk & 31;
  int h      = (blk >> 5) & 15;
  int b      = blk >> 9;
  int qtile  = qgroup * 4 + wslot;
  int s0     = qtile * 16;

  const _Float16* qbase = Qh + (((size_t)b * NB_HEAD + h) * SS + s0) * DH;
  const _Float16* kbase = Kh + ((size_t)b * NB_HEAD + h) * SS * DH;
  const _Float16* vbase = Vt + ((size_t)b * NB_HEAD + h) * DH * SS;

  // q A-fragments (d = 0..31, 32..63); q already scaled by 1/sqrt(Dh)
  v16h aq[2];
#pragma unroll
  for (int c = 0; c < 2; ++c) {
    const _Float16* qr = qbase + (size_t)m * DH + c * 32;
    v8h q0 = *(const v8h*)(qr + 16 * half);
    v8h q1 = *(const v8h*)(qr + 16 * half + 8);
#pragma unroll
    for (int i = 0; i < 8; ++i) { aq[c][i] = q0[i]; aq[c][8 + i] = q1[i]; }
  }

  int vlen = Vlen[b];
  int qlen = Qlen[b];

  float rm[8], rl[8];
#pragma unroll
  for (int r = 0; r < 8; ++r) { rm[r] = -3.0e38f; rl[r] = 0.f; }
  v8f oacc[4] = {};
  _Float16* pw = pbuf[wslot];

  // cooperative stage of K(32x64) + V(64x32) tiles for chunk j into buffer `buf`
  auto stage = [&](int buf, int j) {
    int sk = j * 32;
#pragma unroll
    for (int rpt = 0; rpt < 2; ++rpt) {
      int idx = tid + rpt * 128;               // 0..255
      int krow = idx >> 3, kseg = idx & 7;     // K: 32 rows x 8 x 16B
      copy16_g2l(kbase + (size_t)(sk + krow) * DH + kseg * 8,
                 &kbuf[buf][krow * 64 + kseg * 8]);
      int vrow = idx >> 2, vseg = idx & 3;     // V: 64 rows x 4 x 16B
      copy16_g2l(vbase + (size_t)vrow * SS + sk + vseg * 8,
                 &vbuf[buf][vrow * 32 + vseg * 8]);
    }
  };

  stage(0, 0);
  for (int j = 0; j < SS / 32; ++j) {
    int cur = j & 1;
    int sk = j * 32;
    wait_async_stage();     // my async writes into buf[cur] done (ASYNCcnt==0)
    __syncthreads();        // everyone's writes done; buf[cur^1] free to reuse
    if (j + 1 < SS / 32) stage(cur ^ 1, j + 1);  // overlap next loads w/ compute

    // scores: 2 n-tiles, each = 2 WMMAs over d, B-fragments from LDS
    v8f sc[2];
#pragma unroll
    for (int t = 0; t < 2; ++t) {
      v8f z = {};
#pragma unroll
      for (int c = 0; c < 2; ++c) {
        const _Float16* kr = &kbuf[cur][(t * 16 + m) * 64 + c * 32 + 16 * half];
        v8h b0 = *(const v8h*)(kr);
        v8h b1 = *(const v8h*)(kr + 8);
        v16h bf;
#pragma unroll
        for (int i = 0; i < 8; ++i) { bf[i] = b0[i]; bf[8 + i] = b1[i]; }
        z = wmma_f16(aq[c], bf, z);
      }
      sc[t] = z;
    }
    // additive length mask (matches reference's -1e12)
#pragma unroll
    for (int t = 0; t < 2; ++t)
      if (sk + t * 16 + m >= vlen) {
#pragma unroll
        for (int r = 0; r < 8; ++r) sc[t][r] -= 1.0e12f;
      }
    // row max over 32 cols (reduce across the 16-lane group)
    float mx[8];
#pragma unroll
    for (int r = 0; r < 8; ++r) mx[r] = fmaxf(sc[0][r], sc[1][r]);
#pragma unroll
    for (int d = 1; d < 16; d <<= 1)
#pragma unroll
      for (int r = 0; r < 8; ++r) mx[r] = fmaxf(mx[r], __shfl_xor(mx[r], d, 32));
    // online softmax update
    float corr[8], sum[8];
#pragma unroll
    for (int r = 0; r < 8; ++r) {
      float mnew = fmaxf(rm[r], mx[r]);
      corr[r] = __expf(rm[r] - mnew);
      rm[r] = mnew;
      float p0 = __expf(sc[0][r] - mnew);
      float p1 = __expf(sc[1][r] - mnew);
      sc[0][r] = p0; sc[1][r] = p1;
      sum[r] = p0 + p1;
    }
#pragma unroll
    for (int d = 1; d < 16; d <<= 1)
#pragma unroll
      for (int r = 0; r < 8; ++r) sum[r] += __shfl_xor(sum[r], d, 32);
#pragma unroll
    for (int r = 0; r < 8; ++r) rl[r] = rl[r] * corr[r] + sum[r];
#pragma unroll
    for (int nt = 0; nt < 4; ++nt)
#pragma unroll
      for (int r = 0; r < 8; ++r) oacc[nt][r] *= corr[r];
    // P: accumulator layout -> LDS row-major [16][32] -> A-fragment layout
    // (wave-private buffer; DS ops are in-order within a wave)
#pragma unroll
    for (int t = 0; t < 2; ++t)
#pragma unroll
      for (int r = 0; r < 8; ++r)
        pw[(r + 8 * half) * 32 + t * 16 + m] = (_Float16)sc[t][r];
    v16h ap;
    {
      const _Float16* pr = pw + m * 32;
      v8h p0 = *(const v8h*)(pr + 16 * half);
      v8h p1 = *(const v8h*)(pr + 16 * half + 8);
#pragma unroll
      for (int i = 0; i < 8; ++i) { ap[i] = p0[i]; ap[8 + i] = p1[i]; }
    }
    // O += P @ V  (vbuf is [d][key] so B-fragment loads are contiguous)
#pragma unroll
    for (int nt = 0; nt < 4; ++nt) {
      const _Float16* vr = &vbuf[cur][(nt * 16 + m) * 32 + 16 * half];
      v8h b0 = *(const v8h*)(vr);
      v8h b1 = *(const v8h*)(vr + 8);
      v16h bf;
#pragma unroll
      for (int i = 0; i < 8; ++i) { bf[i] = b0[i]; bf[8 + i] = b1[i]; }
      oacc[nt] = wmma_f16(ap, bf, oacc[nt]);
    }
  }
  // epilogue: out[b][s][h*64 + d] = O/l, zero rows past Q_len
#pragma unroll
  for (int r = 0; r < 8; ++r) {
    int mm = r + 8 * half;
    int s  = s0 + mm;
    float inv  = 1.0f / rl[r];
    bool  live = s < qlen;
#pragma unroll
    for (int nt = 0; nt < 4; ++nt) {
      float val = live ? oacc[nt][r] * inv : 0.0f;
      Out[((size_t)b * SS + s) * DM + h * DH + nt * 16 + m] = val;
    }
  }
}

// ---------------------------------------------------------------------------
extern "C" void kernel_launch(void* const* d_in, const int* in_sizes, int n_in,
                              void* d_out, int out_size, void* d_ws, size_t ws_size,
                              hipStream_t stream) {
  const float* Qseq = (const float*)d_in[0];
  const float* Kseq = (const float*)d_in[1];
  const float* Vseq = (const float*)d_in[2];
  const float* WQ   = (const float*)d_in[3];
  const float* WK   = (const float*)d_in[4];
  const float* WV   = (const float*)d_in[5];
  const int*   Qlen = (const int*)d_in[6];
  const int*   Vlen = (const int*)d_in[7];
  float* out = (float*)d_out;

  const size_t WELE = (size_t)DM * DM;                 // 1M f16 per transposed W
  const size_t TELE = (size_t)BB * NB_HEAD * SS * DH;  // 8M f16 per projected tensor
  _Float16* ws  = (_Float16*)d_ws;
  _Float16* wtq = ws;
  _Float16* wtk = wtq + WELE;
  _Float16* wtv = wtk + WELE;
  _Float16* qh  = wtv + WELE;
  _Float16* kh  = qh + TELE;
  _Float16* vt  = kh + TELE;

  // Pass 0: convert+transpose weights to f16
  {
    dim3 blk(256), grd((unsigned)(WELE / 256));
    wconv_kernel<<<grd, blk, 0, stream>>>(WQ, wtq);
    wconv_kernel<<<grd, blk, 0, stream>>>(WK, wtk);
    wconv_kernel<<<grd, blk, 0, stream>>>(WV, wtv);
  }
  // Pass 1: projections (8192 waves each; 128 thr = 4 waves/block)
  {
    dim3 blk(128), grd(2048);
    proj_kernel<<<grd, blk, 0, stream>>>(Qseq, wtq, qh, 0, 0.125f); // 1/sqrt(64)
    proj_kernel<<<grd, blk, 0, stream>>>(Kseq, wtk, kh, 1, 1.0f);
    proj_kernel<<<grd, blk, 0, stream>>>(Vseq, wtv, vt, 2, 1.0f);
  }
  // Pass 2: flash attention (one block per (b,h,qgroup))
  {
    dim3 blk(128), grd(2048);
    attn_kernel<<<grd, blk, 0, stream>>>(qh, kh, vt, Qlen, Vlen, out);
  }
}
